// CustomTransformerLayer_18803366822025
// MI455X (gfx1250) — matrix operand order repaired and checked by reading, not compiled, not run on hardware
//
#include <hip/hip_runtime.h>
#include <hip/hip_bf16.h>
#include <cmath>

typedef __bf16 bf16;
typedef __attribute__((ext_vector_type(16))) __bf16 v16bf;
typedef __attribute__((ext_vector_type(8)))  float  v8f;
typedef __attribute__((ext_vector_type(4)))  unsigned int v4u;
typedef __attribute__((ext_vector_type(8)))  int v8i;
typedef __attribute__((ext_vector_type(4)))  int v4i;

#define H_DIM   1024
#define NHEAD   16
#define DQK     64
#define DRD     64
#define DVD     128
#define KVLORA  512
#define FFDIM   2816
#define NEXP    16
#define ATT_SCALE 0.08838834764831845f   /* 1/sqrt(64+64) */
#define LN_EPS  1e-5f

// ---------------------------------------------------------------------------
// WMMA helpers (CDNA5 wave32, 16x16x32 bf16 -> f32)
// ---------------------------------------------------------------------------
__device__ __forceinline__ v8f wmma_bf16(v16bf a, v16bf b, v8f c) {
    return __builtin_amdgcn_wmma_f32_16x16x32_bf16(false, a, false, b,
                                                   (short)0, c, false, false);
}

// A-matrix 16x32 bf16 fragment from LDS (row-major [m][k] tile, leading dim ld)
// lanes 0-15: M=lane, K in {0..15}; lanes 16-31: M=lane-16, K in {16..31}
// -> one contiguous 32B run per lane => 2x ds_load_b128
__device__ __forceinline__ v16bf load_a_frag(const bf16* lds, int ld,
                                             int row0, int k0, int lane) {
    int m  = lane & 15;
    int kb = (lane >> 4) << 4;          // 0 or 16
    const bf16* p = lds + (size_t)(row0 + m) * ld + k0;
    v16bf a;
#pragma unroll
    for (int h = 0; h < 8; ++h) a[h]     = p[kb + h];
#pragma unroll
    for (int h = 0; h < 8; ++h) a[8 + h] = p[kb + 8 + h];
    return a;
}

// B-matrix 32x16 bf16 fragment from a TRANSPOSED LDS tile lds[n][k] (ld = k-dim)
// lane n needs B[k0..k0+15][n] = ldsT[n][k0..k0+15]: one contiguous 32B run
// per lane => 2x ds_load_b128
__device__ __forceinline__ v16bf load_bt_frag(const bf16* ldsT, int ld,
                                              int n0, int k0, int lane) {
    int n  = lane & 15;
    int kb = (lane >> 4) << 4;          // 0 or 16
    const bf16* p = ldsT + (size_t)(n0 + n) * ld + k0 + kb;
    v16bf b;
#pragma unroll
    for (int h = 0; h < 16; ++h) b[h] = p[h];
    return b;
}

__device__ __forceinline__ unsigned pack2_bf16(float a, float b) {
    union { bf16 h; unsigned short u; } x, y;
    x.h = (bf16)a; y.h = (bf16)b;
    return (unsigned)x.u | ((unsigned)y.u << 16);
}

// LDS byte offset: generic pointers into the LDS aperture carry the offset in
// the low 32 bits (ISA 10.2).
__device__ __forceinline__ unsigned lds_byte_off(const void* p) {
    return (unsigned)(unsigned long long)p;
}

// ---------------------------------------------------------------------------
// Tensor Data Mover: 2D tile load Global -> LDS.
// D# per CDNA5 ISA, data_size = 8 bytes.
//   tile_d0  : contiguous qwords per row (<= 65535)
//   tile_d1  : number of rows
//   stride_q : row stride in qwords (tensor_dim0_stride)
// Rows are compacted contiguously into LDS starting at lds_off.
// This toolchain exposes the 6-arg builtin (extra int32x8 descriptor group).
// ---------------------------------------------------------------------------
__device__ __forceinline__ void tdm_load_2d(unsigned lds_off, const void* gptr,
                                            unsigned tile_d0, unsigned tile_d1,
                                            unsigned long long stride_q) {
    unsigned long long ga = (unsigned long long)gptr;
    v4u g0;
    g0[0] = 1u;                                          // count=1, user D#
    g0[1] = lds_off;                                     // lds_addr (bytes)
    g0[2] = (unsigned)(ga & 0xFFFFFFFFu);                // global_addr[31:0]
    g0[3] = (unsigned)((ga >> 32) & 0x01FFFFFFu)         // global_addr[56:32]
          | (2u << 30);                                  // type = 2 ("image")
    v8i g1;
    g1[0] = (int)(3u << 16);                             // data_size = 3 (8B)
    g1[1] = (int)((tile_d0 & 0xFFFFu) << 16);            // tensor_dim0[15:0]
    g1[2] = (int)((tile_d0 >> 16) |                      // tensor_dim0[31:16]
                  ((tile_d1 & 0xFFFFu) << 16));          // tensor_dim1[15:0]
    g1[3] = (int)((tile_d1 >> 16) |                      // tensor_dim1[31:16]
                  ((tile_d0 & 0xFFFFu) << 16));          // tile_dim0
    g1[4] = (int)(tile_d1 & 0xFFFFu);                    // tile_dim1 (tile_dim2=0)
    g1[5] = (int)(stride_q & 0xFFFFFFFFu);               // tensor_dim0_stride lo
    g1[6] = (int)((stride_q >> 32) & 0xFFFFu);           // stride hi (dim1_stride=0)
    g1[7] = 0;
    v4i g2 = (v4i){1, 1, 0, 0};                          // tensor_dim2/3 = 1
    v4i g3 = (v4i){0, 0, 0, 0};
    v8i g4 = (v8i){0, 0, 0, 0, 0, 0, 0, 0};              // unused extra group
    __builtin_amdgcn_tensor_load_to_lds(g0, g1, g2, g3, g4, 0);
}

// ---------------------------------------------------------------------------
// LayerNorm (f32 in, bf16 out), generic row stride / dim
// ---------------------------------------------------------------------------
__global__ __launch_bounds__(256) void ln_cast_kernel(
    const float* __restrict__ x, int xstride, int dim,
    const float* __restrict__ g, const float* __restrict__ bb,
    bf16* __restrict__ out, int ostride)
{
    int row = blockIdx.x;
    int tid = threadIdx.x, wave = tid >> 5, lane = tid & 31;
    const float* xr = x + (size_t)row * xstride;
    float sum = 0.f, sq = 0.f;
    for (int i = tid; i < dim; i += 256) { float v = xr[i]; sum += v; sq += v * v; }
#pragma unroll
    for (int off = 16; off; off >>= 1) {
        sum += __shfl_xor(sum, off, 32);
        sq  += __shfl_xor(sq,  off, 32);
    }
    __shared__ float rs[8], rq[8];
    if (lane == 0) { rs[wave] = sum; rq[wave] = sq; }
    __syncthreads();
    if (tid == 0) {
        float a = 0.f, b2 = 0.f;
        for (int w = 0; w < 8; ++w) { a += rs[w]; b2 += rq[w]; }
        rs[0] = a; rq[0] = b2;
    }
    __syncthreads();
    float mean = rs[0] / (float)dim;
    float var  = rq[0] / (float)dim - mean * mean;
    float inv  = rsqrtf(var + LN_EPS);
    bf16* orow = out + (size_t)row * ostride;
    for (int i = tid; i < dim; i += 256)
        orow[i] = (bf16)((xr[i] - mean) * inv * g[i] + bb[i]);
}

// ---------------------------------------------------------------------------
// Generic tiled WMMA GEMM: C[M,N] = act( A[M,K](bf16) @ B[K,N](f32) [+ Res] )
// block tile 128x128, K step 32, 8 waves: 4(M) x 2(N), each wave 32x64.
// A tile staged by the Tensor Data Mover (non-gather) or manual gather path.
// B tile staged TRANSPOSED (sBt[n][k]) with dword-packed bf16 pairs so that
// all fragment reads are ds_load_b128.
// ---------------------------------------------------------------------------
__global__ __launch_bounds__(256) void gemm_wmma_kernel(
    const bf16*  __restrict__ A,  const float* __restrict__ Bw,
    float* Cf, bf16* Cb, const float* __restrict__ Res,
    const int* __restrict__ row_map, const int* __restrict__ tile_expert,
    const int* __restrict__ ntiles, long expert_stride,
    int M, int N, int K, int lda, int ldc, int act)
{
    if (ntiles && (int)blockIdx.y >= *ntiles) return;
    const float* Bp = Bw;
    if (tile_expert) Bp += (long)tile_expert[blockIdx.y] * expert_stride;

    __shared__ bf16 sA[128 * 32];    // [m][k]
    __shared__ bf16 sBt[128 * 32];   // [n][k]  (transposed)

    int tid = threadIdx.x, wave = tid >> 5, lane = tid & 31;
    int bm = blockIdx.y * 128, bn = blockIdx.x * 128;
    int wm = (wave & 3) * 32, wn = (wave >> 2) * 64;
    const bool gather = (row_map != nullptr);
    unsigned sA_off = lds_byte_off(sA);

    // hoisted gather: each thread stages 2 A row-slices, rows fixed per thread
    const bf16* aptr[2] = { nullptr, nullptr };
    int ar[2] = {0, 0}, acol[2] = {0, 0};
    bool aval[2] = { false, false };
    if (gather) {
#pragma unroll
        for (int i = 0; i < 2; ++i) {
            int g = tid * 2 + i;
            ar[i] = g >> 2; acol[i] = (g & 3) * 8;
            int grow = bm + ar[i];
            int src = (grow < M) ? row_map[grow] : -1;
            aval[i] = (src >= 0) && (grow < M);
            if (src < 0) src = 0;
            aptr[i] = A + (size_t)src * lda + acol[i];
        }
    }

    v8f acc[2][4];
#pragma unroll
    for (int i = 0; i < 2; ++i)
#pragma unroll
        for (int j = 0; j < 4; ++j) acc[i][j] = (v8f){0.f,0.f,0.f,0.f,0.f,0.f,0.f,0.f};

    for (int k0 = 0; k0 < K; k0 += 32) {
        __syncthreads();
        // ---- stage A tile (128x32 bf16)
        if (!gather) {
            // TDM 2D tile: 8 qwords/row (64B), 128 rows, row stride lda/4 qwords
            if (wave == 0)
                tdm_load_2d(sA_off, (const void*)(A + (size_t)bm * lda + k0),
                            8u, 128u, (unsigned long long)(lda >> 2));
        } else {
#pragma unroll
            for (int i = 0; i < 2; ++i) {
                uint4 val = make_uint4(0u, 0u, 0u, 0u);
                if (aval[i]) val = *(const uint4*)(aptr[i] + k0);
                *(uint4*)(sA + ar[i] * 32 + acol[i]) = val;
            }
        }
        // ---- stage B tile transposed: 2 tasks/thread, each = 2 k-rows x 4 cols
#pragma unroll
        for (int i = 0; i < 2; ++i) {
            int g  = tid + 256 * i;           // 0..511
            int kp = g >> 5;                  // k-pair 0..15
            int cg = (g & 31) * 4;            // col group
            int cn = bn + cg;
            const float* b0 = Bp + (size_t)(k0 + 2 * kp) * N;
            const float* b1 = b0 + N;
            float4 r0 = make_float4(0.f, 0.f, 0.f, 0.f);
            float4 r1 = r0;
            if (cn + 3 < N) {
                r0 = *(const float4*)(b0 + cn);
                r1 = *(const float4*)(b1 + cn);
            } else {
                if (cn + 0 < N) { r0.x = b0[cn + 0]; r1.x = b1[cn + 0]; }
                if (cn + 1 < N) { r0.y = b0[cn + 1]; r1.y = b1[cn + 1]; }
                if (cn + 2 < N) { r0.z = b0[cn + 2]; r1.z = b1[cn + 2]; }
                if (cn + 3 < N) { r0.w = b0[cn + 3]; r1.w = b1[cn + 3]; }
            }
            int kw = 2 * kp;                  // even k index
            *(unsigned*)(sBt + (cg + 0) * 32 + kw) = pack2_bf16(r0.x, r1.x);
            *(unsigned*)(sBt + (cg + 1) * 32 + kw) = pack2_bf16(r0.y, r1.y);
            *(unsigned*)(sBt + (cg + 2) * 32 + kw) = pack2_bf16(r0.z, r1.z);
            *(unsigned*)(sBt + (cg + 3) * 32 + kw) = pack2_bf16(r0.w, r1.w);
        }
        if (k0 + 32 < K)
            __builtin_prefetch(Bp + (size_t)(k0 + 32) * N + bn, 0, 0);
        if (!gather && wave == 0)
            __builtin_amdgcn_s_wait_tensorcnt(0);
        __syncthreads();

        v16bf af[2], bfv[4];
#pragma unroll
        for (int i = 0; i < 2; ++i) af[i]  = load_a_frag(sA, 32, wm + i * 16, 0, lane);
#pragma unroll
        for (int j = 0; j < 4; ++j) bfv[j] = load_bt_frag(sBt, 32, wn + j * 16, 0, lane);
#pragma unroll
        for (int i = 0; i < 2; ++i)
#pragma unroll
            for (int j = 0; j < 4; ++j)
                acc[i][j] = wmma_bf16(af[i], bfv[j], acc[i][j]);
    }

    // ---- epilogue: C element (lane,r) -> m = r + 8*(lane/16), n = lane%16
    int nn = lane & 15, moff = (lane >> 4) * 8;
#pragma unroll
    for (int i = 0; i < 2; ++i)
#pragma unroll
        for (int j = 0; j < 4; ++j)
#pragma unroll
            for (int r = 0; r < 8; ++r) {
                int row = bm + wm + i * 16 + moff + r;
                int col = bn + wn + j * 16 + nn;
                if (row < M && col < N) {
                    float v = acc[i][j][r];
                    if (Res) v += Res[(size_t)row * ldc + col];
                    if (act == 1) v = v / (1.f + __expf(-v));   // SiLU
                    if (Cf) Cf[(size_t)row * ldc + col] = v;
                    if (Cb) Cb[(size_t)row * ldc + col] = (bf16)v;
                }
            }
}

// ---------------------------------------------------------------------------
// RoPE + Q/K/V assembly.  q:[tok][2048] kvd:[tok][576] kvu:[tok][3072]
// -> Qb/Kb/Vb bf16 [B][NH][S][128]
// ---------------------------------------------------------------------------
__global__ __launch_bounds__(128) void rope_assemble_kernel(
    const float* __restrict__ q, const float* __restrict__ kvd,
    const float* __restrict__ kvu,
    bf16* __restrict__ Qb, bf16* __restrict__ Kb, bf16* __restrict__ Vb, int S)
{
    int s = blockIdx.x, h = blockIdx.y, b = blockIdx.z;
    int d = threadIdx.x;                               // 0..127
    size_t tok = (size_t)b * S + s;
    const float* qr  = q   + tok * (NHEAD * 128) + h * 128;
    const float* kur = kvu + tok * (NHEAD * 192) + h * 192;
    const float* kpe = kvd + tok * (KVLORA + DRD) + KVLORA;

    float qv, kv;
    if (d < DQK) { qv = qr[d]; kv = kur[d]; }
    else {
        int j = d - DQK;
        int i = (j < 32) ? j : j - 32;
        float ang = (float)s * __expf(-0.2878231366f * (float)i); // 10000^(-2i/64)
        float c = __cosf(ang), sn = __sinf(ang);
        float q1 = qr[DQK + i], q2 = qr[DQK + 32 + i];
        float k1 = kpe[i],      k2 = kpe[32 + i];
        if (j < 32) { qv = q1 * c - q2 * sn; kv = k1 * c - k2 * sn; }
        else        { qv = q2 * c + q1 * sn; kv = k2 * c + k1 * sn; }
    }
    float vv = kur[DQK + d];                           // v is 128 wide
    size_t o = (((size_t)b * NHEAD + h) * S + s) * 128 + d;
    Qb[o] = (bf16)qv; Kb[o] = (bf16)kv; Vb[o] = (bf16)vv;
}

// ---------------------------------------------------------------------------
// Flash attention (WMMA, online softmax).
// grid.x = S/128 (q tile), grid.y = B*NH.  8 waves, each owns 16 q rows.
// Q tile and K chunks staged by the Tensor Data Mover (contiguous bf16 tiles);
// V staged transposed (sVt[d][key]) via dword packing. sK[key][d] doubles as
// the transposed layout for the scores B-matrix (K^T). All fragment reads are
// contiguous (ds_load_b128 pairs).
// ---------------------------------------------------------------------------
__global__ __launch_bounds__(256) void attn_wmma_kernel(
    const bf16* __restrict__ Q, const bf16* __restrict__ Kk,
    const bf16* __restrict__ V, const float* __restrict__ mask,
    bf16* __restrict__ ctx, int S)
{
    __shared__ bf16 sQ[128 * 128];     // [q][d]
    __shared__ bf16 sK[32 * 128];      // [key][d]
    __shared__ bf16 sVt[128 * 32];     // [d][key]  (transposed)
    __shared__ bf16 sP[8][16 * 32];    // per-wave P tile [m][key]

    int tid = threadIdx.x, wave = tid >> 5, lane = tid & 31;
    int q0 = blockIdx.x * 128;
    int bh = blockIdx.y;
    const bf16* Qp = Q  + (size_t)bh * S * 128;
    const bf16* Kp = Kk + (size_t)bh * S * 128;
    const bf16* Vp = V  + (size_t)bh * S * 128;
    unsigned sQ_off = lds_byte_off(sQ);
    unsigned sK_off = lds_byte_off(sK);

    // stage Q tile 128x128 bf16 = 32KB contiguous, via TDM
    if (wave == 0) {
        tdm_load_2d(sQ_off, (const void*)(Qp + (size_t)q0 * 128), 4096u, 1u, 4096ull);
        __builtin_amdgcn_s_wait_tensorcnt(0);
    }
    __syncthreads();

    v16bf qf[4];
#pragma unroll
    for (int kk = 0; kk < 4; ++kk)
        qf[kk] = load_a_frag(sQ, 128, wave * 16, kk * 32, lane);

    v8f acc[8];
#pragma unroll
    for (int j = 0; j < 8; ++j) acc[j] = (v8f){0.f,0.f,0.f,0.f,0.f,0.f,0.f,0.f};
    float run_m[8], run_l[8];
#pragma unroll
    for (int r = 0; r < 8; ++r) { run_m[r] = -1e30f; run_l[r] = 0.f; }

    int nn = lane & 15, moff = (lane >> 4) * 8;

    for (int kc = 0; kc < S; kc += 32) {
        __syncthreads();
        // ---- stage K chunk 32x128 bf16 = 8KB contiguous, via TDM
        if (wave == 0)
            tdm_load_2d(sK_off, (const void*)(Kp + (size_t)kc * 128), 1024u, 1u, 1024ull);
        // ---- stage V chunk transposed: 1 task/thread = 2 keys x 8 d
        {
            int e2 = (tid >> 4) * 2;          // even key index 0..30
            int dg = (tid & 15) * 8;          // d base
            uint4 r0 = *(const uint4*)(Vp + (size_t)(kc + e2) * 128 + dg);
            uint4 r1 = *(const uint4*)(Vp + (size_t)(kc + e2 + 1) * 128 + dg);
            unsigned w0[4] = { r0.x, r0.y, r0.z, r0.w };
            unsigned w1[4] = { r1.x, r1.y, r1.z, r1.w };
#pragma unroll
            for (int w = 0; w < 4; ++w) {
                int d0 = dg + 2 * w;
                unsigned lo = (w0[w] & 0xFFFFu) | (w1[w] << 16);
                unsigned hi = (w0[w] >> 16) | (w1[w] & 0xFFFF0000u);
                *(unsigned*)(sVt + (size_t)d0 * 32 + e2)       = lo;
                *(unsigned*)(sVt + (size_t)(d0 + 1) * 32 + e2) = hi;
            }
        }
        if (wave == 0) __builtin_amdgcn_s_wait_tensorcnt(0);
        __syncthreads();

        // ---- scores: 16x32 = two 16x16 C tiles, contraction over d=128
        v8f s0 = (v8f){0.f,0.f,0.f,0.f,0.f,0.f,0.f,0.f};
        v8f s1 = s0;
#pragma unroll
        for (int kk = 0; kk < 4; ++kk) {
            v16bf b0 = load_bt_frag(sK, 128, 0,  kk * 32, lane);
            v16bf b1 = load_bt_frag(sK, 128, 16, kk * 32, lane);
            s0 = wmma_bf16(qf[kk], b0, s0);
            s1 = wmma_bf16(qf[kk], b1, s1);
        }

        // ---- online softmax (row reduction within 16-lane group = C layout)
#pragma unroll
        for (int r = 0; r < 8; ++r) {
            int qrow = q0 + wave * 16 + moff + r;
            const float* mrow = mask + (size_t)qrow * S + kc;
            float v0 = s0[r] * ATT_SCALE + mrow[nn];
            float v1 = s1[r] * ATT_SCALE + mrow[16 + nn];
            float mx = fmaxf(v0, v1);
#pragma unroll
            for (int off = 8; off; off >>= 1)
                mx = fmaxf(mx, __shfl_xor(mx, off, 32));
            float nm = fmaxf(run_m[r], mx);
            float sc = __expf(run_m[r] - nm);
            float p0 = __expf(v0 - nm);
            float p1 = __expf(v1 - nm);
            float ps = p0 + p1;
#pragma unroll
            for (int off = 8; off; off >>= 1)
                ps += __shfl_xor(ps, off, 32);
            run_l[r] = run_l[r] * sc + ps;
            run_m[r] = nm;
#pragma unroll
            for (int j = 0; j < 8; ++j) acc[j][r] *= sc;
            sP[wave][(moff + r) * 32 + nn]      = (bf16)p0;
            sP[wave][(moff + r) * 32 + 16 + nn] = (bf16)p1;
        }
        // same-wave LDS store -> cross-lane load; LDS is in-order, keep the
        // compiler from reordering and make the dependency explicit.
        asm volatile("s_wait_dscnt 0x0" ::: "memory");

        v16bf pf = load_a_frag(sP[wave], 32, 0, 0, lane);
#pragma unroll
        for (int j = 0; j < 8; ++j) {
            v16bf vf = load_bt_frag(sVt, 32, j * 16, 0, lane);
            acc[j] = wmma_bf16(pf, vf, acc[j]);
        }
    }

    // normalize + write ctx (B,S,NH*128)
    int b = bh / NHEAD, h = bh % NHEAD;
#pragma unroll
    for (int r = 0; r < 8; ++r) {
        float inv = 1.f / run_l[r];
        int srow = q0 + wave * 16 + moff + r;
        size_t base = ((size_t)b * S + srow) * (NHEAD * DVD) + (size_t)h * DVD;
#pragma unroll
        for (int j = 0; j < 8; ++j)
            ctx[base + j * 16 + nn] = (bf16)(acc[j][r] * inv);
    }
}

// ---------------------------------------------------------------------------
// Router: sigmoid(h2 @ Wr + br), top-2 (ties -> lower index), normalized.
// ---------------------------------------------------------------------------
__global__ __launch_bounds__(32) void router_kernel(
    const bf16* __restrict__ h2, const float* __restrict__ rw,
    const float* __restrict__ rb, int* __restrict__ topi,
    float* __restrict__ topv, int* __restrict__ counts)
{
    int tok = blockIdx.x;
    int e = threadIdx.x;
    __shared__ float p[NEXP];
    if (e < NEXP) {
        float acc = rb[e];
        const bf16* hr = h2 + (size_t)tok * H_DIM;
        for (int k = 0; k < H_DIM; ++k)
            acc += (float)hr[k] * rw[(size_t)k * NEXP + e];
        p[e] = 1.f / (1.f + __expf(-acc));
    }
    __syncthreads();
    if (threadIdx.x == 0) {
        int i1 = 0; float v1 = p[0];
        for (int i = 1; i < NEXP; ++i) if (p[i] > v1) { v1 = p[i]; i1 = i; }
        int i2 = (i1 == 0) ? 1 : 0; float v2 = p[i2];
        for (int i = 0; i < NEXP; ++i)
            if (i != i1 && p[i] > v2) { v2 = p[i]; i2 = i; }
        float s = v1 + v2;
        topi[tok * 2 + 0] = i1; topi[tok * 2 + 1] = i2;
        topv[tok * 2 + 0] = v1 / s; topv[tok * 2 + 1] = v2 / s;
        atomicAdd(&counts[i1], 1);
        atomicAdd(&counts[i2], 1);
    }
}

__global__ void moe_init_kernel(int* counts, int* fill, int* slot_token, int nslot)
{
    int i = blockIdx.x * blockDim.x + threadIdx.x;
    if (i < nslot) slot_token[i] = -1;
    if (i < NEXP) { counts[i] = 0; fill[i] = 0; }
}

// single-thread exclusive scan with 128-row padding + tile->expert map
__global__ void moe_scan_kernel(const int* counts, int* slot_base,
                                int* tile_expert, int* ntiles)
{
    if (threadIdx.x == 0 && blockIdx.x == 0) {
        int t = 0;
        for (int e = 0; e < NEXP; ++e) {
            slot_base[e] = t * 128;
            int nt = (counts[e] + 127) / 128;
            for (int i = 0; i < nt; ++i) tile_expert[t + i] = e;
            t += nt;
        }
        *ntiles = t;
    }
}

__global__ void moe_scatter_kernel(const int* topi, const float* topv,
                                   int* fill, const int* slot_base,
                                   int* slot_token, float* slot_w,
                                   int* slot_of, int ntok)
{
    int tok = blockIdx.x * blockDim.x + threadIdx.x;
    if (tok >= ntok) return;
#pragma unroll
    for (int j = 0; j < 2; ++j) {
        int e = topi[tok * 2 + j];
        int pos = atomicAdd(&fill[e], 1);
        int slot = slot_base[e] + pos;
        slot_token[slot] = tok;
        slot_w[slot] = topv[tok * 2 + j];
        slot_of[tok * 2 + j] = slot;
    }
}

// out[tok][c] = h_res + w0 * fc2(slot0) + w1 * fc2(slot1)   (deterministic)
__global__ __launch_bounds__(256) void moe_combine_kernel(
    const float* __restrict__ hres, const float* __restrict__ inter2,
    const int* __restrict__ slot_of, const float* __restrict__ slot_w,
    float* __restrict__ out, int ntok)
{
    long idx = (long)blockIdx.x * blockDim.x + threadIdx.x;
    if (idx >= (long)ntok * H_DIM) return;
    int tok = (int)(idx / H_DIM);
    int c   = (int)(idx - (long)tok * H_DIM);
    int s0 = slot_of[tok * 2 + 0], s1 = slot_of[tok * 2 + 1];
    float v = hres[idx]
            + slot_w[s0] * inter2[(size_t)s0 * H_DIM + c]
            + slot_w[s1] * inter2[(size_t)s1 * H_DIM + c];
    out[idx] = v;
}

// ---------------------------------------------------------------------------
// Host-side orchestration
// ---------------------------------------------------------------------------
extern "C" void kernel_launch(void* const* d_in, const int* in_sizes, int n_in,
                              void* d_out, int out_size, void* d_ws, size_t ws_size,
                              hipStream_t stream)
{
    const float* hidden = (const float*)d_in[0];
    const float* amask  = (const float*)d_in[1];
    const float* ln1_s  = (const float*)d_in[2];
    const float* ln1_b  = (const float*)d_in[3];
    const float* q_w    = (const float*)d_in[4];
    const float* kvd_w  = (const float*)d_in[5];
    const float* kvln_s = (const float*)d_in[6];
    const float* kvln_b = (const float*)d_in[7];
    const float* kvu_w  = (const float*)d_in[8];
    const float* out_w  = (const float*)d_in[9];
    const float* ln2_s  = (const float*)d_in[10];
    const float* ln2_b  = (const float*)d_in[11];
    const float* rou_w  = (const float*)d_in[12];
    const float* rou_b  = (const float*)d_in[13];
    const float* fc1_w  = (const float*)d_in[14];
    const float* fc2_w  = (const float*)d_in[15];

    const int S   = (int)(sqrtf((float)in_sizes[1]) + 0.5f);
    const int TOK = in_sizes[0] / H_DIM;
    const int B   = TOK / S;
    const int NSLOT = ((2 * TOK) / 128 + NEXP) * 128;   // padded slot capacity
    const int NTILE_MAX = NSLOT / 128;

    char* w = (char*)d_ws;
    auto alloc = [&](size_t bytes) -> void* {
        void* p = (void*)w;
        w += (bytes + 255) & ~(size_t)255;
        return p;
    };
    bf16*  h1_bf   = (bf16*)alloc((size_t)TOK * H_DIM * 2);
    float* q_f32   = (float*)alloc((size_t)TOK * (NHEAD * 128) * 4);
    float* kvd_f32 = (float*)alloc((size_t)TOK * (KVLORA + DRD) * 4);
    bf16*  kvc_bf  = (bf16*)alloc((size_t)TOK * KVLORA * 2);
    float* kvu_f32 = (float*)alloc((size_t)TOK * (NHEAD * 192) * 4);
    bf16*  Qb      = (bf16*)alloc((size_t)B * NHEAD * S * 128 * 2);
    bf16*  Kb      = (bf16*)alloc((size_t)B * NHEAD * S * 128 * 2);
    bf16*  Vb      = (bf16*)alloc((size_t)B * NHEAD * S * 128 * 2);
    bf16*  ctx_bf  = (bf16*)alloc((size_t)TOK * (NHEAD * DVD) * 2);
    float* h_res   = (float*)alloc((size_t)TOK * H_DIM * 4);
    bf16*  h2_bf   = (bf16*)alloc((size_t)TOK * H_DIM * 2);
    int*   topi    = (int*)alloc((size_t)TOK * 2 * 4);
    float* topv    = (float*)alloc((size_t)TOK * 2 * 4);
    int*   counts  = (int*)alloc(NEXP * 4);
    int*   fill    = (int*)alloc(NEXP * 4);
    int*   sbase   = (int*)alloc(NEXP * 4);
    int*   texp    = (int*)alloc((size_t)NTILE_MAX * 4);
    int*   ntl     = (int*)alloc(4);
    int*   stok    = (int*)alloc((size_t)NSLOT * 4);
    float* sw      = (float*)alloc((size_t)NSLOT * 4);
    int*   sof     = (int*)alloc((size_t)TOK * 2 * 4);
    bf16*  inter   = (bf16*)alloc((size_t)NSLOT * FFDIM * 2);
    float* inter2  = (float*)alloc((size_t)NSLOT * H_DIM * 4);
    (void)ws_size; (void)n_in; (void)out_size;

    // 1) LN1 -> bf16
    ln_cast_kernel<<<TOK, 256, 0, stream>>>(hidden, H_DIM, H_DIM,
                                            ln1_s, ln1_b, h1_bf, H_DIM);
    // 2) q = h1 @ q_w   (4096x1024 @ 1024x2048)
    gemm_wmma_kernel<<<dim3((NHEAD * 128) / 128, TOK / 128), 256, 0, stream>>>(
        h1_bf, q_w, q_f32, nullptr, nullptr, nullptr, nullptr, nullptr, 0,
        TOK, NHEAD * 128, H_DIM, H_DIM, NHEAD * 128, 0);
    // 3) kvd = h1 @ kv_down_w  (N=576, partial tile)
    gemm_wmma_kernel<<<dim3((KVLORA + DRD + 127) / 128, TOK / 128), 256, 0, stream>>>(
        h1_bf, kvd_w, kvd_f32, nullptr, nullptr, nullptr, nullptr, nullptr, 0,
        TOK, KVLORA + DRD, H_DIM, H_DIM, KVLORA + DRD, 0);
    // 4) kv LN on first 512 columns -> bf16
    ln_cast_kernel<<<TOK, 256, 0, stream>>>(kvd_f32, KVLORA + DRD, KVLORA,
                                            kvln_s, kvln_b, kvc_bf, KVLORA);
    // 5) kvu = kvc @ kv_up_w  (4096x512 @ 512x3072)
    gemm_wmma_kernel<<<dim3((NHEAD * 192) / 128, TOK / 128), 256, 0, stream>>>(
        kvc_bf, kvu_w, kvu_f32, nullptr, nullptr, nullptr, nullptr, nullptr, 0,
        TOK, NHEAD * 192, KVLORA, KVLORA, NHEAD * 192, 0);
    // 6) RoPE + assemble Q/K/V
    rope_assemble_kernel<<<dim3(S, NHEAD, B), 128, 0, stream>>>(
        q_f32, kvd_f32, kvu_f32, Qb, Kb, Vb, S);
    // 7) flash attention
    attn_wmma_kernel<<<dim3(S / 128, B * NHEAD), 256, 0, stream>>>(
        Qb, Kb, Vb, amask, ctx_bf, S);
    // 8) h = hidden + ctx @ out_w
    gemm_wmma_kernel<<<dim3(H_DIM / 128, TOK / 128), 256, 0, stream>>>(
        ctx_bf, out_w, h_res, nullptr, hidden, nullptr, nullptr, nullptr, 0,
        TOK, H_DIM, NHEAD * DVD, NHEAD * DVD, H_DIM, 0);
    // 9) LN2 -> bf16
    ln_cast_kernel<<<TOK, 256, 0, stream>>>(h_res, H_DIM, H_DIM,
                                            ln2_s, ln2_b, h2_bf, H_DIM);
    // 10) router + top2 + slot building
    moe_init_kernel<<<(NSLOT + 255) / 256, 256, 0, stream>>>(counts, fill, stok, NSLOT);
    router_kernel<<<TOK, 32, 0, stream>>>(h2_bf, rou_w, rou_b, topi, topv, counts);
    moe_scan_kernel<<<1, 32, 0, stream>>>(counts, sbase, texp, ntl);
    moe_scatter_kernel<<<(TOK + 255) / 256, 256, 0, stream>>>(
        topi, topv, fill, sbase, stok, sw, sof, TOK);
    // 11) grouped fc1: inter[slot] = silu(h2[token(slot)] @ fc1_w[e])
    gemm_wmma_kernel<<<dim3(FFDIM / 128, NTILE_MAX), 256, 0, stream>>>(
        h2_bf, fc1_w, nullptr, inter, nullptr, stok, texp, ntl,
        (long)H_DIM * FFDIM, NSLOT, FFDIM, H_DIM, H_DIM, FFDIM, 1);
    // 12) grouped fc2: inter2[slot] = inter[slot] @ fc2_w[e]
    gemm_wmma_kernel<<<dim3(H_DIM / 128, NTILE_MAX), 256, 0, stream>>>(
        inter, fc2_w, inter2, nullptr, nullptr, nullptr, texp, ntl,
        (long)FFDIM * H_DIM, NSLOT, H_DIM, FFDIM, FFDIM, H_DIM, 0);
    // 13) combine + residual -> d_out
    moe_combine_kernel<<<(int)(((long)TOK * H_DIM + 255) / 256), 256, 0, stream>>>(
        h_res, inter2, sof, sw, (float*)d_out, TOK);
}
